// LIF_79293686219008
// MI455X (gfx1250) — matrix operand order repaired and checked by reading, not compiled, not run on hardware
//
#include <hip/hip_runtime.h>

#define NIN    256
#define NREC   1024
#define TSTEPS 500
#define BATCH  64
#define THRV   0.4f
#define DECAYV 0.95122942450071400910f   // exp(-1/20)

typedef __attribute__((ext_vector_type(16))) __bf16 v16bf;
typedef __attribute__((ext_vector_type(8)))  __bf16 v8bf;
typedef __attribute__((ext_vector_type(4)))  __bf16 v4bf;
typedef __attribute__((ext_vector_type(8)))  float  v8f;
typedef __attribute__((ext_vector_type(4)))  float  v4f;

// ---------------------------------------------------------------------------
// Prologue: convert weights to bf16 and transpose to N-major [n][k] so the
// WMMA B-fragment (contiguous K order used here: lane n=l&15, K = 16*(l>>4)+0..15)
// is one contiguous 32-byte load per lane.
// ---------------------------------------------------------------------------
__global__ void lif_prep_weights(const float* __restrict__ w_in,
                                 const float* __restrict__ w_rec,
                                 __bf16* __restrict__ w_inT,
                                 __bf16* __restrict__ w_recT) {
  int idx = blockIdx.x * blockDim.x + threadIdx.x;
  const int NE_IN  = NIN * NREC;    // 262144
  const int NE_REC = NREC * NREC;   // 1048576
  if (idx < NE_IN) {
    int n = idx / NIN, k = idx % NIN;
    w_inT[idx] = (__bf16)w_in[(size_t)k * NREC + n];
  } else if (idx < NE_IN + NE_REC) {
    int j = idx - NE_IN;
    int n = j / NREC, k = j % NREC;
    w_recT[j] = (__bf16)w_rec[(size_t)k * NREC + n];
  }
}

// A-fragment for 16-bit A 16x32 (lane m=l&15; K = 16*half + 0..15,
// matching the B-fragment): two 16B LDS loads + concat.
__device__ inline v16bf load_afrag(const __bf16* p) {
  v8bf lo = *(const v8bf*)p;
  v8bf hi = *(const v8bf*)(p + 8);
  return __builtin_shufflevector(lo, hi, 0, 1, 2, 3, 4, 5, 6, 7,
                                         8, 9, 10, 11, 12, 13, 14, 15);
}

// ---------------------------------------------------------------------------
// Main scan kernel: blockIdx.x = batch slab of 16 rows (recurrence is
// per-sample independent, so no cross-workgroup sync is ever needed).
// 32 waves per workgroup; wave w owns neuron columns [32w, 32w+32).
// Spikes z are exchanged between waves via LDS (bf16) each step.
// Weights are streamed from L2 every step; the laundered zero offset below
// stops LLVM from hoisting the (loop-invariant) weight loads and spilling
// them, while keeping the pointers traceable to global addr space
// (global_load_b128 with LOADcnt only, instead of flat_load + DScnt).
// ---------------------------------------------------------------------------
__launch_bounds__(1024)
__global__ void lif_scan_kernel(const float* __restrict__ x,
                                const __bf16* __restrict__ w_inT,
                                const __bf16* __restrict__ w_recT,
                                float* __restrict__ out) {
  constexpr int ZP = NREC + 8;  // +8 bf16 row pad -> conflict-free ds_load_b128
  constexpr int XP = NIN + 8;
  __shared__ __align__(16) __bf16 z_lds[16][ZP];   // ~33 KB
  __shared__ __align__(16) __bf16 x_lds[16][XP];   // ~8.4 KB

  const int tid   = threadIdx.x;
  const int lane  = tid & 31;
  const int wave  = tid >> 5;       // 0..31
  const int b0    = blockIdx.x * 16;

  const int khalf = lane >> 4;      // 0/1
  const int l15   = lane & 15;
  const int kb8   = khalf * 8;      // unused; the A-frag also uses kb16
  const int kb16  = khalf * 16;     // B-frag K offset within a 32-K tile

  const int ncol0 = wave * 32;
  const int ncol1 = ncol0 + 16;

  const __bf16* pB_in0  = w_inT  + (size_t)(ncol0 + l15) * NIN  + kb16;
  const __bf16* pB_in1  = w_inT  + (size_t)(ncol1 + l15) * NIN  + kb16;
  const __bf16* pB_rec0 = w_recT + (size_t)(ncol0 + l15) * NREC + kb16;
  const __bf16* pB_rec1 = w_recT + (size_t)(ncol1 + l15) * NREC + kb16;

  float* out_z = out;
  float* out_v = out + (size_t)BATCH * TSTEPS * NREC;

  // z(t=-1) = 0
  for (int i = tid; i < 16 * ZP; i += 1024)
    (&z_lds[0][0])[i] = (__bf16)0.0f;

  v8f v0 = {}, v1 = {};   // membrane potentials (C-fragment layout)
  v8f z0 = {}, z1 = {};   // previous spikes (0/1)

  for (int t = 0; t < TSTEPS; ++t) {
    // Opaque zero: makes the weight-load addresses loop-variant to the
    // optimizer (no hoisting/spilling) without hiding their address space.
    int off = 0;
    asm volatile("" : "+v"(off));
    const __bf16* qin0  = pB_in0  + off;
    const __bf16* qin1  = pB_in1  + off;
    const __bf16* qrec0 = pB_rec0 + off;
    const __bf16* qrec1 = pB_rec1 + off;

    // Stage x_t (16 x 256 f32 -> bf16 LDS): 4 elements per thread.
    {
      int row = tid >> 6;            // 0..15
      int c4  = (tid & 63) << 2;     // 0..252
      v4f xv = *(const v4f*)(x + ((size_t)(b0 + row) * TSTEPS + t) * NIN + c4);
      v4bf xb = { (__bf16)xv.x, (__bf16)xv.y, (__bf16)xv.z, (__bf16)xv.w };
      *(v4bf*)&x_lds[row][c4] = xb;
    }
    __syncthreads();   // x_lds staged; z_lds writes from step t-1 visible

    v8f acc0 = {}, acc1 = {};

    // v_in += x_t @ w_in   (K = 256)
    const __bf16* pa = &x_lds[l15][kb16];
    #pragma unroll
    for (int kt = 0; kt < NIN / 32; ++kt) {
      v16bf a   = load_afrag(pa + kt * 32);
      v16bf bf0 = *(const v16bf*)(qin0 + kt * 32);
      v16bf bf1 = *(const v16bf*)(qin1 + kt * 32);
      acc0 = __builtin_amdgcn_wmma_f32_16x16x32_bf16(false, a, false, bf0,
                                                     (short)0, acc0, false, false);
      acc1 = __builtin_amdgcn_wmma_f32_16x16x32_bf16(false, a, false, bf1,
                                                     (short)0, acc1, false, false);
    }

    // v_in += z @ w_rec   (K = 1024)
    const __bf16* pz = &z_lds[l15][kb16];
    #pragma unroll 8
    for (int kt = 0; kt < NREC / 32; ++kt) {
      v16bf a   = load_afrag(pz + kt * 32);
      v16bf bf0 = *(const v16bf*)(qrec0 + kt * 32);
      v16bf bf1 = *(const v16bf*)(qrec1 + kt * 32);
      acc0 = __builtin_amdgcn_wmma_f32_16x16x32_bf16(false, a, false, bf0,
                                                     (short)0, acc0, false, false);
      acc1 = __builtin_amdgcn_wmma_f32_16x16x32_bf16(false, a, false, bf1,
                                                     (short)0, acc1, false, false);
    }

    // LIF update: new_v = DECAY*v + v_in - z*THR; new_z = (new_v > THR)
    #pragma unroll
    for (int r = 0; r < 8; ++r) {
      float nv0 = DECAYV * v0[r] + acc0[r] - z0[r] * THRV;
      float nv1 = DECAYV * v1[r] + acc1[r] - z1[r] * THRV;
      v0[r] = nv0;
      v1[r] = nv1;
      z0[r] = nv0 > THRV ? 1.0f : 0.0f;
      z1[r] = nv1 > THRV ? 1.0f : 0.0f;
    }

    __syncthreads();   // all waves finished reading z_lds for this step

    // Publish new spikes to LDS + stream z, v to global (nontemporal: keep
    // the L2 for the resident bf16 weights).
    #pragma unroll
    for (int r = 0; r < 8; ++r) {
      int m = khalf * 8 + r;                 // C-fragment row
      z_lds[m][ncol0 + l15] = (__bf16)z0[r];
      z_lds[m][ncol1 + l15] = (__bf16)z1[r];
      size_t o = ((size_t)(b0 + m) * TSTEPS + t) * NREC + l15;
      __builtin_nontemporal_store(z0[r], out_z + o + ncol0);
      __builtin_nontemporal_store(z1[r], out_z + o + ncol1);
      __builtin_nontemporal_store(v0[r], out_v + o + ncol0);
      __builtin_nontemporal_store(v1[r], out_v + o + ncol1);
    }
  }
}

extern "C" void kernel_launch(void* const* d_in, const int* in_sizes, int n_in,
                              void* d_out, int out_size, void* d_ws, size_t ws_size,
                              hipStream_t stream) {
  const float* x     = (const float*)d_in[0];  // [64, 500, 256]
  const float* w_in  = (const float*)d_in[1];  // [256, 1024]
  const float* w_rec = (const float*)d_in[2];  // [1024, 1024]

  __bf16* w_inT  = (__bf16*)d_ws;                      // 512 KB
  __bf16* w_recT = w_inT + (size_t)NIN * NREC;         // 2 MB

  const int total = NIN * NREC + NREC * NREC;          // 1,310,720
  lif_prep_weights<<<(total + 255) / 256, 256, 0, stream>>>(w_in, w_rec,
                                                           w_inT, w_recT);
  lif_scan_kernel<<<BATCH / 16, 1024, 0, stream>>>(x, w_inT, w_recT,
                                                   (float*)d_out);
}
